// RAN_M_5222680232620
// MI455X (gfx1250) — matrix operand order repaired and checked by reading, not compiled, not run on hardware
//
#include <hip/hip_runtime.h>
#include <hip/hip_bf16.h>

typedef __attribute__((ext_vector_type(16))) _Float16 v16h;
typedef __attribute__((ext_vector_type(8)))  _Float16 v8h;
typedef __attribute__((ext_vector_type(8)))  float    v8f;

#define B_     32
#define NOPE   2048
#define NMAC   1024
#define DIN    64
#define KOUT   128
#define OSPLIT 8
#define MTILE  128
#define NEG_SLOPE 0.2f
#define MASK_FILL (-9e10f)

__device__ __forceinline__ float lrelu(float x) { return x >= 0.f ? x : NEG_SLOPE * x; }

// ---------------------------------------------------------------------------
// wt[d]      = sum_k ope_w[k][d] * ope_alpha[k]
// wt[64+d]   = sum_k mac_w[k][d] * mac_alpha[k]
// (attn = (feat @ W^T) @ alpha  ==  feat @ (W^T alpha) -- collapse the 128-dim)
// ---------------------------------------------------------------------------
__global__ void k_wt(const float* __restrict__ ope_w, const float* __restrict__ mac_w,
                     const float* __restrict__ ope_a, const float* __restrict__ mac_a,
                     float* __restrict__ wt) {
    int d = threadIdx.x;                 // 64 threads
    float s0 = 0.f, s1 = 0.f;
    for (int k = 0; k < KOUT; ++k) {
        s0 += ope_w[k * DIN + d] * ope_a[k];
        s1 += mac_w[k * DIN + d] * mac_a[k];
    }
    wt[d] = s0;
    wt[64 + d] = s1;
}

// ---------------------------------------------------------------------------
// attn[r] = dot(feat[r, :64], wt)
// ---------------------------------------------------------------------------
__global__ void k_attn(const float* __restrict__ feat, const float* __restrict__ wt,
                       float* __restrict__ attn, int nrows) {
    int r = blockIdx.x * blockDim.x + threadIdx.x;
    if (r >= nrows) return;
    const float* f = feat + (size_t)r * DIN;
    float s = 0.f;
#pragma unroll
    for (int d = 0; d < DIN; ++d) s += f[d] * wt[d];
    attn[r] = s;
}

// ---------------------------------------------------------------------------
// Shared helper: stage W[128][64] as f16 into LDS (one conversion per block,
// instead of one private conversion per lane).
// ---------------------------------------------------------------------------
__device__ __forceinline__ void stage_weights(const float* __restrict__ w, _Float16* wl) {
    const int t = threadIdx.x;                  // 256 threads: k = t/2, half = (t&1)*32
    const float* wr = w + (size_t)(t >> 1) * DIN + (t & 1) * 32;
    _Float16* dl = wl + (t >> 1) * DIN + (t & 1) * 32;
#pragma unroll
    for (int i = 0; i < 32; ++i) dl[i] = (_Float16)wr[i];
}

// ---------------------------------------------------------------------------
// h_ope^T (f16): hT[b][k][o] = sum_d feat_ope[b][o][d] * ope_w[k][d]
// One wave per 16-o strip, 8 k-tiles per wave. f16 WMMA, f32 accumulate.
// A layout (16-bit 16x32): lane half h: v0..3 = K 8h+{0..7}, v4..7 = K 16+8h+{0..7}
// B layout (16-bit 32x16): lane half h: v0..3 = K 8h+{0..7}, v4..7 = K 16+8h+{0..7}
// ---------------------------------------------------------------------------
__global__ void __launch_bounds__(256) k_proj(const float* __restrict__ feat,
                                              const float* __restrict__ w,
                                              _Float16* __restrict__ hT) {
    __shared__ _Float16 wl[KOUT * DIN];          // 16 KB
    stage_weights(w, wl);

    const int b    = blockIdx.y;
    const int wave = threadIdx.x >> 5;
    const int lane = threadIdx.x & 31;
    const int h    = lane >> 4;
    const int ln   = lane & 15;
    const int obase = (blockIdx.x * 8 + wave) * 16;
    const int row   = obase + ln;

    const float* fr = feat + ((size_t)b * NOPE + row) * DIN;
    v16h a0, a1;
#pragma unroll
    for (int i = 0; i < 8; ++i) {
        a0[i]     = (_Float16)fr[8 * h + i];
        a0[i + 8] = (_Float16)fr[16 + 8 * h + i];
        a1[i]     = (_Float16)fr[32 + 8 * h + i];
        a1[i + 8] = (_Float16)fr[48 + 8 * h + i];
    }
    __syncthreads();

#pragma unroll
    for (int kt = 0; kt < 8; ++kt) {
        const int k = kt * 16 + ln;
        const _Float16* wr = wl + k * DIN;
        v16h b0, b1;
#pragma unroll
        for (int i = 0; i < 16; ++i) {
            b0[i] = wr[8 * h + i + (i & 8)];
            b1[i] = wr[32 + 8 * h + i + (i & 8)];
        }
        v8f c = {};
        c = __builtin_amdgcn_wmma_f32_16x16x32_f16(false, a0, false, b0, (short)0, c, false, false);
        c = __builtin_amdgcn_wmma_f32_16x16x32_f16(false, a1, false, b1, (short)0, c, false, false);
        v8h st;
#pragma unroll
        for (int r = 0; r < 8; ++r) st[r] = (_Float16)c[r];
        // C elem r = (M=r+8h, N=ln): 8 consecutive o -> one 16B store per lane
        *(v8h*)(hT + ((size_t)(b * KOUT + k)) * NOPE + obase + 8 * h) = st;
    }
}

// ---------------------------------------------------------------------------
// out[b][m][k] = h_res = sum_d feat_mac[b][m][d] * res_w[k][d]   (f32 init of out)
// ---------------------------------------------------------------------------
__global__ void __launch_bounds__(256) k_res(const float* __restrict__ feat,
                                             const float* __restrict__ w,
                                             float* __restrict__ out) {
    __shared__ _Float16 wl[KOUT * DIN];          // 16 KB
    stage_weights(w, wl);

    const int b    = blockIdx.y;
    const int wave = threadIdx.x >> 5;
    const int lane = threadIdx.x & 31;
    const int h    = lane >> 4;
    const int ln   = lane & 15;
    const int mbase = (blockIdx.x * 8 + wave) * 16;
    const int row   = mbase + ln;

    const float* fr = feat + ((size_t)b * NMAC + row) * DIN;
    v16h a0, a1;
#pragma unroll
    for (int i = 0; i < 8; ++i) {
        a0[i]     = (_Float16)fr[8 * h + i];
        a0[i + 8] = (_Float16)fr[16 + 8 * h + i];
        a1[i]     = (_Float16)fr[32 + 8 * h + i];
        a1[i + 8] = (_Float16)fr[48 + 8 * h + i];
    }
    __syncthreads();

#pragma unroll
    for (int kt = 0; kt < 8; ++kt) {
        const int k = kt * 16 + ln;
        const _Float16* wr = wl + k * DIN;
        v16h b0, b1;
#pragma unroll
        for (int i = 0; i < 16; ++i) {
            b0[i] = wr[8 * h + i + (i & 8)];
            b1[i] = wr[32 + 8 * h + i + (i & 8)];
        }
        v8f c = {};
        c = __builtin_amdgcn_wmma_f32_16x16x32_f16(false, a0, false, b0, (short)0, c, false, false);
        c = __builtin_amdgcn_wmma_f32_16x16x32_f16(false, a1, false, b1, (short)0, c, false, false);
#pragma unroll
        for (int r = 0; r < 8; ++r)
            out[((size_t)(b * NMAC + mbase + 8 * h + r)) * KOUT + k] = c[r];
    }
}

// ---------------------------------------------------------------------------
// Online softmax stats over o, split into OSPLIT partial ranges of 256.
// part[(b*NMAC+m)*OSPLIT+os] = {max, sumexp}
// ---------------------------------------------------------------------------
__global__ void __launch_bounds__(256) k_stats(const int* __restrict__ adj,
                                               const int* __restrict__ bidx,
                                               const float* __restrict__ attn_ope,
                                               const float* __restrict__ attn_mac,
                                               float* __restrict__ part) {
    const int b  = blockIdx.z;
    const int os = blockIdx.y;
    const int m  = blockIdx.x * 256 + threadIdx.x;
    const int g  = bidx[b];
    const float cm = attn_mac[b * NMAC + m];
    const int* arow = adj + ((size_t)g * NOPE + os * 256) * NMAC + m;
    const float* ao = attn_ope + b * NOPE + os * 256;

    float mx = -3.0e38f, sum = 0.f;
    for (int o = 0; o < 256; ++o) {
        float e = (arow[(size_t)o * NMAC] == 1) ? lrelu(ao[o] + cm) : MASK_FILL;
        if (e > mx) { sum = sum * __expf(mx - e) + 1.f; mx = e; }
        else        { sum += __expf(e - mx); }
    }
    float* p = part + ((size_t)(b * NMAC + m) * OSPLIT + os) * 2;
    p[0] = mx;
    p[1] = sum;
}

// stats[idx] = {max over splits, 1/sum}
__global__ void k_reduce(const float* __restrict__ part, float* __restrict__ stats) {
    int idx = blockIdx.x * blockDim.x + threadIdx.x;   // b*NMAC + m
    const float* p = part + (size_t)idx * OSPLIT * 2;
    float mx = -3.0e38f;
#pragma unroll
    for (int i = 0; i < OSPLIT; ++i) mx = fmaxf(mx, p[2 * i]);
    float s = 0.f;
#pragma unroll
    for (int i = 0; i < OSPLIT; ++i) s += p[2 * i + 1] * __expf(p[2 * i] - mx);
    stats[idx * 2]     = mx;
    stats[idx * 2 + 1] = 1.f / s;
}

// ---------------------------------------------------------------------------
// out[b][m][k] += sum_o alpha[b][o][m] * h_ope[b][o][k]
// Block: 256 thr = 8 waves; m-tile = 128 (8 subtiles of 16 per wave), each wave
// owns one 16-wide k-tile. alpha regenerated per 32-o chunk into LDS (f16),
// B operand loaded contiguously from transposed h_ope; B reused over 8 WMMAs.
// Mask tile for chunk+2 is prefetched (global_prefetch_b8).
// ---------------------------------------------------------------------------
__global__ void __launch_bounds__(256) k_out(const int* __restrict__ adj,
                                             const int* __restrict__ bidx,
                                             const float* __restrict__ attn_ope,
                                             const float* __restrict__ attn_mac,
                                             const float* __restrict__ stats,
                                             const _Float16* __restrict__ hT,
                                             float* __restrict__ out) {
    __shared__ _Float16 atile[MTILE * 48];       // [m][o] rows padded to 48 halves (96B)

    const int b     = blockIdx.y;
    const int mtile = blockIdx.x * MTILE;
    const int g     = bidx[b];
    const int t     = threadIdx.x;

    // alpha-generation role: thread -> (m = t%128, o strip = t/128 * 16 .. +15)
    const int   mloc  = t & (MTILE - 1);
    const int   ohalf = t >> 7;                  // 0..1
    const int   m     = mtile + mloc;
    const float cm    = attn_mac[b * NMAC + m];
    const float Mv    = stats[(b * NMAC + m) * 2];
    const float Sinv  = stats[(b * NMAC + m) * 2 + 1];
    const int*   abase = adj + (size_t)g * NOPE * NMAC + mtile;   // column-tile base
    const float* ao    = attn_ope + b * NOPE;

    // WMMA role
    const int lane = t & 31, wave = t >> 5;
    const int h = lane >> 4, ln = lane & 15;
    const int k = wave * 16 + ln;
    const _Float16* hrow = hT + ((size_t)(b * KOUT + k)) * NOPE;

    v8f acc[8] = {v8f{}, v8f{}, v8f{}, v8f{}, v8f{}, v8f{}, v8f{}, v8f{}};

    for (int ch = 0; ch < NOPE / 32; ++ch) {
        const int obase = ch * 32;

        // prefetch mask tile two chunks ahead: 32 o-rows x 512B, 256 threads
        if (obase + 64 < NOPE) {
            const int pre_o = obase + 64 + (t & 31);
            __builtin_prefetch(abase + (size_t)pre_o * NMAC + ((t >> 5) * 16), 0, 1);
        }

#pragma unroll
        for (int i = 0; i < 16; ++i) {
            const int ol = ohalf * 16 + i;
            const int o  = obase + ol;
            float e  = (abase[(size_t)o * NMAC + mloc] == 1) ? lrelu(ao[o] + cm) : MASK_FILL;
            float al = __expf(e - Mv) * Sinv;
            atile[mloc * 48 + ol] = (_Float16)al;
        }
        __syncthreads();

        // B: K = 16h + {0..15} -> 32B contiguous from transposed h_ope row
        v16h bv = *(const v16h*)(hrow + obase + 16 * h);

#pragma unroll
        for (int s = 0; s < 8; ++s) {
            const _Float16* ar = atile + (s * 16 + ln) * 48;
            v8h lo = *(const v8h*)(ar + 16 * h);       // K 16h+{0..7}
            v8h hi = *(const v8h*)(ar + 16 * h + 8);   // K 16h+8+{0..7}
            v16h av;
#pragma unroll
            for (int i = 0; i < 8; ++i) { av[i] = lo[i]; av[i + 8] = hi[i]; }
            acc[s] = __builtin_amdgcn_wmma_f32_16x16x32_f16(false, av, false, bv,
                                                            (short)0, acc[s], false, false);
        }
        __syncthreads();
    }

    // accumulate into out (already holds h_res)
    float* ob = out + (size_t)b * NMAC * KOUT + k;
#pragma unroll
    for (int s = 0; s < 8; ++s) {
#pragma unroll
        for (int r = 0; r < 8; ++r) {
            const int m2 = mtile + s * 16 + 8 * h + r;
            float* p = ob + (size_t)m2 * KOUT;
            *p += acc[s][r];
        }
    }
}

// ---------------------------------------------------------------------------
extern "C" void kernel_launch(void* const* d_in, const int* in_sizes, int n_in,
                              void* d_out, int out_size, void* d_ws, size_t ws_size,
                              hipStream_t stream) {
    (void)in_sizes; (void)n_in; (void)out_size; (void)ws_size;
    const int*   adj   = (const int*)  d_in[0];   // [64][2048][1024]
    const int*   bidx  = (const int*)  d_in[1];   // [32]
    const float* fope  = (const float*)d_in[2];   // [32][2048][64]
    const float* fmac  = (const float*)d_in[3];   // [32][1024][64]
    const float* ope_w = (const float*)d_in[4];   // [128][64]
    const float* mac_w = (const float*)d_in[5];   // [128][64]
    const float* ope_a = (const float*)d_in[6];   // [128][1]
    const float* mac_a = (const float*)d_in[7];   // [128][1]
    const float* res_w = (const float*)d_in[8];   // [128][64]
    float* out = (float*)d_out;                   // [32][1024][128] f32

    char* ws = (char*)d_ws;                       // ~18.7 MB used
    _Float16* hT    = (_Float16*)(ws + 0);              // 16,777,216 B
    float* attn_ope = (float*)(ws + 16777216);          //    262,144 B
    float* attn_mac = (float*)(ws + 17039360);          //    131,072 B
    float* wt       = (float*)(ws + 17170432);          //        512 B
    float* part     = (float*)(ws + 17170944);          //  2,097,152 B
    float* stats    = (float*)(ws + 19268096);          //    262,144 B

    k_wt   <<<1, 64, 0, stream>>>(ope_w, mac_w, ope_a, mac_a, wt);
    k_attn <<<(B_ * NOPE) / 256, 256, 0, stream>>>(fope, wt,      attn_ope, B_ * NOPE);
    k_attn <<<(B_ * NMAC) / 256, 256, 0, stream>>>(fmac, wt + 64, attn_mac, B_ * NMAC);
    k_proj <<<dim3(NOPE / 128, B_), 256, 0, stream>>>(fope, ope_w, hT);
    k_res  <<<dim3(NMAC / 128, B_), 256, 0, stream>>>(fmac, res_w, out);
    k_stats<<<dim3(NMAC / 256, OSPLIT, B_), 256, 0, stream>>>(adj, bidx, attn_ope, attn_mac, part);
    k_reduce<<<(B_ * NMAC) / 256, 256, 0, stream>>>(part, stats);
    k_out  <<<dim3(NMAC / MTILE, B_), 256, 0, stream>>>(adj, bidx, attn_ope, attn_mac, stats, hT, out);
}
